// PolygonGATV2Encoder_9740985827988
// MI455X (gfx1250) — compile-verified, matrix operand order checked by reading, not run on hardware
//
#include <hip/hip_runtime.h>
#include <hip/hip_bf16.h>
#include <math.h>

typedef __attribute__((ext_vector_type(16))) _Float16 v16h;
typedef __attribute__((ext_vector_type(8)))  _Float16 v8h;
typedef __attribute__((ext_vector_type(8)))  float    v8f;

#define HID 32

// ---------- helpers ----------
__device__ __forceinline__ unsigned f2ord(float f) {
    unsigned u = __float_as_uint(f);
    return (u & 0x80000000u) ? ~u : (u | 0x80000000u);
}
__device__ __forceinline__ float ord2f(unsigned u) {
    return __uint_as_float((u & 0x80000000u) ? (u & 0x7FFFFFFFu) : ~u);
}
__device__ __forceinline__ float wave_sum32(float v) {
    #pragma unroll
    for (int m = 16; m > 0; m >>= 1) v += __shfl_xor(v, m, 32);
    return v;
}

// ---------- fills ----------
__global__ void fillf_k(float* p, float v, int n) {
    int t = blockIdx.x * blockDim.x + threadIdx.x;
    if (t < n) p[t] = v;
}
__global__ void fillu_k(unsigned* p, unsigned v, int n) {
    int t = blockIdx.x * blockDim.x + threadIdx.x;
    if (t < n) p[t] = v;
}

// ---------- conversions ----------
__global__ void cvt16_k(const float* __restrict__ a, _Float16* __restrict__ b, int n) {
    int t = blockIdx.x * blockDim.x + threadIdx.x;
    if (t < n) b[t] = (_Float16)a[t];
}
// W (K x OUT) fp32 row-major -> Wt (OUT x K) f16 row-major (i.e. column-major B)
__global__ void wt16_k(const float* __restrict__ W, _Float16* __restrict__ Wt, int K, int OUT) {
    int t = blockIdx.x * blockDim.x + threadIdx.x;
    if (t >= K * OUT) return;
    int k = t / OUT, o = t - k * OUT;
    Wt[(size_t)o * K + k] = (_Float16)W[t];
}

// ---------- input linear: h = x(N x 8) @ W(8 x 32) + b ----------
__global__ void inlin_k(const float* __restrict__ x, const float* __restrict__ W,
                        const float* __restrict__ b, float* __restrict__ h, int N_) {
    __shared__ float sW[8 * 32];
    __shared__ float sb[32];
    for (int i = threadIdx.x; i < 256; i += blockDim.x) sW[i] = W[i];
    if (threadIdx.x < 32) sb[threadIdx.x] = b[threadIdx.x];
    __syncthreads();
    int t = blockIdx.x * blockDim.x + threadIdx.x;
    if (t >= N_ * 32) return;
    int n = t >> 5, j = t & 31;
    const float* xr = x + (size_t)n * 8;
    float s = sb[j];
    #pragma unroll
    for (int k = 0; k < 8; ++k) s += xr[k] * sW[k * 32 + j];
    h[t] = s;
}

// ---------- WMMA f16 GEMM: C(M x OUT) = A(M x K) * B(K x OUT) + bias ----------
// A: f16 row-major. Bt: f16, OUT x K row-major (B column-major).
// One wave computes one 16x16 tile; v_wmma_f32_16x16x32_f16 over K/32 steps.
__global__ void gemm16_k(const _Float16* __restrict__ A, const _Float16* __restrict__ Bt,
                         const float* __restrict__ bias, float* __restrict__ C,
                         int M, int K, int OUT) {
    int gtid = blockIdx.x * blockDim.x + threadIdx.x;
    int wave = gtid >> 5;
    int lane = threadIdx.x & 31;
    int tilesN = OUT >> 4;
    int tilesM = M >> 4;
    if (wave >= tilesM * tilesN) return;
    int tm = wave / tilesN;
    int tn = wave - tm * tilesN;
    int l16 = lane & 15;
    int kb = (lane >> 4) << 3;  // lanes 0-15: K base 0 ; lanes 16-31: K base 8
    const _Float16* arow = A  + (size_t)(tm * 16 + l16) * K;
    const _Float16* brow = Bt + (size_t)(tn * 16 + l16) * K;
    union HV { v16h v; v8h h[2]; };
    v8f acc = {};
    for (int k0 = 0; k0 < K; k0 += 32) {
        HV a, b;
        a.h[0] = *(const v8h*)(arow + k0 + kb);
        a.h[1] = *(const v8h*)(arow + k0 + kb + 16);
        b.h[0] = *(const v8h*)(brow + k0 + kb);
        b.h[1] = *(const v8h*)(brow + k0 + kb + 16);
        acc = __builtin_amdgcn_wmma_f32_16x16x32_f16(
            false, a.v, false, b.v, (short)0, acc, false, false);
    }
    int col = tn * 16 + l16;
    float bv = bias[col];
    int rbase = tm * 16 + ((lane >> 4) << 3);
    #pragma unroll
    for (int r = 0; r < 8; ++r)
        C[(size_t)(rbase + r) * OUT + col] = acc[r] + bv;
}

// ---------- edge pass A: alpha + segment max ----------
__global__ void edge_alpha_k(const int* __restrict__ ei, const float* __restrict__ ea,
                             const float* __restrict__ xl, const float* __restrict__ xr,
                             const float* __restrict__ We, const float* __restrict__ att,
                             float* __restrict__ alpha, unsigned* __restrict__ maxb,
                             int E_, int H, int hshift) {
    __shared__ float sWe[4 * 128];
    __shared__ float sAtt[128];
    int HC = H << 5;
    for (int i = threadIdx.x; i < 4 * HC; i += blockDim.x) sWe[i] = We[i];
    for (int i = threadIdx.x; i < HC; i += blockDim.x) sAtt[i] = att[i];
    __syncthreads();
    int t = blockIdx.x * blockDim.x + threadIdx.x;
    if (t >= E_ * H) return;
    int e = t >> hshift, h = t & (H - 1);
    int src = ei[e], dst = ei[E_ + e];
    float4 a4 = *(const float4*)(ea + (size_t)e * 4);
    const float* pl = xl + (size_t)src * HC + (h << 5);
    const float* pr = xr + (size_t)dst * HC + (h << 5);
    const float* w0 = sWe + (h << 5);
    const float* at = sAtt + (h << 5);
    float al = 0.f;
    #pragma unroll
    for (int c = 0; c < 32; c += 4) {
        float4 l4 = *(const float4*)(pl + c);
        float4 r4 = *(const float4*)(pr + c);
        float la[4] = {l4.x, l4.y, l4.z, l4.w};
        float ra[4] = {r4.x, r4.y, r4.z, r4.w};
        #pragma unroll
        for (int i = 0; i < 4; ++i) {
            float ev = a4.x * w0[c + i] + a4.y * w0[HC + c + i] +
                       a4.z * w0[2 * HC + c + i] + a4.w * w0[3 * HC + c + i];
            float m = la[i] + ra[i] + ev;
            m = m > 0.f ? m : 0.2f * m;     // leaky_relu 0.2
            al += m * at[c + i];
        }
    }
    alpha[t] = al;
    atomicMax(maxb + (size_t)dst * H + h, f2ord(al));
}

// ---------- edge pass B: exp-weighted scatter ----------
__global__ void edge_accum_k(const int* __restrict__ ei, const float* __restrict__ xl,
                             const float* __restrict__ alpha, const unsigned* __restrict__ maxb,
                             float* __restrict__ den, float* __restrict__ numb,
                             int E_, int H, int hshift) {
    int t = blockIdx.x * blockDim.x + threadIdx.x;
    if (t >= E_ * H) return;
    int e = t >> hshift, h = t & (H - 1);
    int src = ei[e], dst = ei[E_ + e];
    int HC = H << 5;
    float mx = ord2f(maxb[(size_t)dst * H + h]);
    float ex = __expf(alpha[t] - mx);
    atomicAdd(den + (size_t)dst * H + h, ex);
    const float* pl = xl + (size_t)src * HC + (h << 5);
    float* pn = numb + (size_t)dst * HC + (h << 5);
    #pragma unroll
    for (int c = 0; c < 32; c += 4) {
        float4 v = *(const float4*)(pl + c);
        atomicAdd(pn + c + 0, v.x * ex);
        atomicAdd(pn + c + 1, v.y * ex);
        atomicAdd(pn + c + 2, v.z * ex);
        atomicAdd(pn + c + 3, v.w * ex);
    }
}

// ---------- node finalize: div + bias + residual + LayerNorm + GELU ----------
// one wave (32 lanes) per node; lane owns channels oc = lane + 32*j (head == j)
__global__ void fin_k(const float* __restrict__ numb, const float* __restrict__ den,
                      const float* __restrict__ hin, const float* __restrict__ resW,
                      const float* __restrict__ resb, const float* __restrict__ bias,
                      const float* __restrict__ lng, const float* __restrict__ lnb,
                      float* __restrict__ hout, int N_, int H, int D, int IN, int useRes) {
    __shared__ float sW[4096];  // max IN*D for residual layers = 32*128 = 4096
    if (useRes) {
        int kw = IN * D;
        for (int i = threadIdx.x; i < kw; i += blockDim.x) sW[i] = resW[i];
    }
    __syncthreads();
    int wave = blockIdx.x * (blockDim.x >> 5) + (threadIdx.x >> 5);
    int lane = threadIdx.x & 31;
    if (wave >= N_) return;
    int V = D >> 5;  // channels per lane: 4 (D=128) or 1 (D=32)
    float tv[4];
    size_t nb = (size_t)wave * D;
    for (int j = 0; j < V; ++j) {
        int oc = lane + (j << 5);
        float dn = den[(size_t)wave * H + j];
        tv[j] = numb[nb + oc] / (dn + 1e-16f) + bias[oc];
    }
    if (useRes) {
        float racc[4];
        for (int j = 0; j < V; ++j) racc[j] = resb[lane + (j << 5)];
        const float* hrow = hin + (size_t)wave * IN;
        for (int k = 0; k < IN; ++k) {
            float hk = hrow[k];
            for (int j = 0; j < V; ++j) racc[j] += hk * sW[k * D + lane + (j << 5)];
        }
        for (int j = 0; j < V; ++j) tv[j] += racc[j];
    } else {
        const float* hrow = hin + (size_t)wave * IN;  // IN == D here
        for (int j = 0; j < V; ++j) tv[j] += hrow[lane + (j << 5)];
    }
    float s = 0.f;
    for (int j = 0; j < V; ++j) s += tv[j];
    s = wave_sum32(s);
    float mu = s / (float)D;
    float vv = 0.f;
    for (int j = 0; j < V; ++j) { float d = tv[j] - mu; vv += d * d; }
    vv = wave_sum32(vv);
    float rs = rsqrtf(vv / (float)D + 1e-5f);
    for (int j = 0; j < V; ++j) {
        int oc = lane + (j << 5);
        float y = lng[oc] * (tv[j] - mu) * rs + lnb[oc];
        hout[nb + oc] = 0.5f * y * (1.f + erff(y * 0.70710678118654752f));  // exact gelu
    }
}

// ---------- sigmoid-gated pooling ----------
__global__ void pool_k(const float* __restrict__ h, const float* __restrict__ pW,
                       const float* __restrict__ pb, const int* __restrict__ batch,
                       float* __restrict__ pooled, int N_) {
    int wave = blockIdx.x * (blockDim.x >> 5) + (threadIdx.x >> 5);
    int lane = threadIdx.x & 31;
    if (wave >= N_) return;
    float v = h[(size_t)wave * 32 + lane];
    float p = wave_sum32(v * pW[lane]) + pb[0];
    float att = 1.f / (1.f + __expf(-p));
    atomicAdd(pooled + (size_t)batch[wave] * 32 + lane, v * att);
}

// ---------- final MLP + L2 normalize (one block per graph) ----------
__global__ void mlp_k(const float* __restrict__ pooled, const float* __restrict__ W1,
                      const float* __restrict__ b1, const float* __restrict__ W2,
                      const float* __restrict__ b2, float* __restrict__ out) {
    __shared__ float sp[32];
    __shared__ float se[128];
    __shared__ float red[128];
    int g = blockIdx.x, t = threadIdx.x;
    if (t < 32) sp[t] = pooled[(size_t)g * 32 + t];
    __syncthreads();
    float a = b1[t];
    #pragma unroll
    for (int k = 0; k < 32; ++k) a += sp[k] * W1[k * 128 + t];
    se[t] = fmaxf(a, 0.f);
    __syncthreads();
    float b = b2[t];
    for (int k = 0; k < 128; ++k) b += se[k] * W2[k * 128 + t];
    red[t] = b * b;
    __syncthreads();
    for (int s = 64; s > 0; s >>= 1) {
        if (t < s) red[t] += red[t + s];
        __syncthreads();
    }
    float nrm = fmaxf(sqrtf(red[0]), 1e-12f);
    out[(size_t)g * 128 + t] = b / nrm;
}

extern "C" void kernel_launch(void* const* d_in, const int* in_sizes, int n_in,
                              void* d_out, int out_size, void* d_ws, size_t ws_size,
                              hipStream_t stream) {
    (void)in_sizes; (void)n_in; (void)out_size; (void)ws_size;
    const int N = 100000, E = 1600000, G = 1024;

    const float* x      = (const float*)d_in[0];
    const int*   ei     = (const int*)  d_in[1];
    const float* ea     = (const float*)d_in[2];
    const int*   batch  = (const int*)  d_in[3];
    const float* in_W   = (const float*)d_in[4];
    const float* in_b   = (const float*)d_in[5];
    const float* pool_W = (const float*)d_in[6];
    const float* pool_b = (const float*)d_in[7];
    const float* out1_W = (const float*)d_in[8];
    const float* out1_b = (const float*)d_in[9];
    const float* out2_W = (const float*)d_in[10];
    const float* out2_b = (const float*)d_in[11];

    // layer param base indices (insertion order; layers 0 and 3 carry resW/resb)
    const int lbase[4]  = {12, 23, 32, 41};
    const int hasRes[4] = {1, 0, 0, 1};
    const int INd[4]    = {32, 128, 128, 128};
    const int Hh[4]     = {4, 4, 4, 1};
    const int hsh[4]    = {2, 2, 2, 0};
    const int Dd[4]     = {128, 128, 128, 32};

    // workspace carve-up
    char* w = (char*)d_ws;
    size_t off = 0;
    auto carve = [&](size_t bytes) -> void* {
        void* p = w + off;
        off += (bytes + 255) & ~(size_t)255;
        return p;
    };
    float*     h0     = (float*)    carve((size_t)N * 128 * 4);
    float*     h1     = (float*)    carve((size_t)N * 128 * 4);
    _Float16*  hf     = (_Float16*) carve((size_t)N * 128 * 2);
    float*     xl     = (float*)    carve((size_t)N * 128 * 4);
    float*     xr     = (float*)    carve((size_t)N * 128 * 4);
    float*     numb   = (float*)    carve((size_t)N * 128 * 4);
    float*     alpha  = (float*)    carve((size_t)E * 4 * 4);
    unsigned*  maxb   = (unsigned*) carve((size_t)N * 4 * 4);
    float*     den    = (float*)    carve((size_t)N * 4 * 4);
    _Float16*  WlT    = (_Float16*) carve((size_t)128 * 128 * 2);
    _Float16*  WrT    = (_Float16*) carve((size_t)128 * 128 * 2);
    float*     pooled = (float*)    carve((size_t)G * 32 * 4);

    const int B = 256;
    auto nb = [](int n, int b) { return (n + b - 1) / b; };

    // input projection: h0 = x @ in_W + in_b
    inlin_k<<<nb(N * 32, B), B, 0, stream>>>(x, in_W, in_b, h0, N);

    float* hc = h0;
    float* hn = h1;
    for (int L = 0; L < 4; ++L) {
        int IN = INd[L], H = Hh[L], D = Dd[L];
        const float* Wl   = (const float*)d_in[lbase[L] + 0];
        const float* bl   = (const float*)d_in[lbase[L] + 1];
        const float* Wr   = (const float*)d_in[lbase[L] + 2];
        const float* br   = (const float*)d_in[lbase[L] + 3];
        const float* We   = (const float*)d_in[lbase[L] + 4];
        const float* att  = (const float*)d_in[lbase[L] + 5];
        const float* bias = (const float*)d_in[lbase[L] + 6];
        const float* lng  = (const float*)d_in[lbase[L] + 7];
        const float* lnb  = (const float*)d_in[lbase[L] + 8];
        const float* resW = hasRes[L] ? (const float*)d_in[lbase[L] + 9]  : nullptr;
        const float* resb = hasRes[L] ? (const float*)d_in[lbase[L] + 10] : nullptr;

        // h -> f16, weights -> transposed f16
        cvt16_k<<<nb(N * IN, B), B, 0, stream>>>(hc, hf, N * IN);
        wt16_k<<<nb(IN * D, B), B, 0, stream>>>(Wl, WlT, IN, D);
        wt16_k<<<nb(IN * D, B), B, 0, stream>>>(Wr, WrT, IN, D);

        // WMMA node projections: x_l = h@Wl+bl ; x_r = h@Wr+br
        int tiles = (N / 16) * (D / 16);
        gemm16_k<<<nb(tiles, 8), B, 0, stream>>>(hf, WlT, bl, xl, N, IN, D);
        gemm16_k<<<nb(tiles, 8), B, 0, stream>>>(hf, WrT, br, xr, N, IN, D);

        // softmax scratch init
        fillu_k<<<nb(N * H, B), B, 0, stream>>>(maxb, 0u, N * H);
        fillf_k<<<nb(N * H, B), B, 0, stream>>>(den, 0.f, N * H);
        fillf_k<<<nb(N * D, B), B, 0, stream>>>(numb, 0.f, N * D);

        // edge passes
        int tE = E * H;
        edge_alpha_k<<<nb(tE, B), B, 0, stream>>>(ei, ea, xl, xr, We, att, alpha, maxb, E, H, hsh[L]);
        edge_accum_k<<<nb(tE, B), B, 0, stream>>>(ei, xl, alpha, maxb, den, numb, E, H, hsh[L]);

        // finalize node state (8 nodes per 256-thread block)
        fin_k<<<nb(N, 8), B, 0, stream>>>(numb, den, hc, resW, resb, bias, lng, lnb,
                                          hn, N, H, D, IN, hasRes[L]);
        float* tmp = hc; hc = hn; hn = tmp;
    }

    // pooling + output MLP
    fillf_k<<<nb(G * 32, B), B, 0, stream>>>(pooled, 0.f, G * 32);
    pool_k<<<nb(N, 8), B, 0, stream>>>(hc, pool_W, pool_b, batch, pooled, N);
    mlp_k<<<G, 128, 0, stream>>>(pooled, out1_W, out1_b, out2_W, out2_b, (float*)d_out);
}
